// TnlAttention_28260884807953
// MI455X (gfx1250) — matrix operand order repaired and checked by reading, not compiled, not run on hardware
//
#include <hip/hip_runtime.h>
#include <stdint.h>

// Problem constants (match reference)
#define B_  2
#define N_  2048
#define D_  2048
#define H_  16
#define HD_ 128
#define C_  128
#define NC_ (N_ / C_)        // 16
#define MTOT_ (B_ * N_)      // 4096

// Row-major bf16 tile stride (ushorts): 136 -> 272B rows (16B aligned, bank-spread)
#define LDA_ 136

typedef __attribute__((ext_vector_type(16))) __bf16 v16bf;
typedef __attribute__((ext_vector_type(8)))  float  v8f;

union FragU { v16bf v; unsigned short u[16]; unsigned int d[8]; uint4 q[2]; };
union Pack4 { uint4 q; unsigned int w[4]; unsigned short us[8]; };

__device__ inline unsigned short f2bf(float f) {
  unsigned int u = __builtin_bit_cast(unsigned int, f);
  u += 0x7fffu + ((u >> 16) & 1u);
  return (unsigned short)(u >> 16);
}
__device__ inline float bf2f(unsigned short s) {
  unsigned int u = ((unsigned int)s) << 16;
  return __builtin_bit_cast(float, u);
}

__device__ inline v8f vzero8() {
  v8f z;
#pragma unroll
  for (int i = 0; i < 8; ++i) z[i] = 0.0f;
  return z;
}

__device__ inline v8f wmma_bf16(v16bf a, v16bf b, v8f c) {
#if defined(__HIP_DEVICE_COMPILE__)
  return __builtin_amdgcn_wmma_f32_16x16x32_bf16(false, a, false, b, (short)0, c, false, false);
#else
  return c;
#endif
}

// A fragment (16x32 MxK bf16) from row-major LDS tile: two ds_load_b128.
// half=0 lane: row dwords {0..7}; half=1 lane: {8..15}.
// Requires ld % 8 == 0 and 16B-aligned base.
__device__ inline v16bf load_frag_a128(const unsigned short* p, int ld) {
  int lane = threadIdx.x & 31;
  int m = lane & 15, half = lane >> 4;
  const uint4* q = (const uint4*)(p + m * ld);
  FragU f;
  f.q[0] = q[2 * half];
  f.q[1] = q[2 * half + 1];
  return f.v;
}

// A fragment with per-K scaling (only for the small U-GEMM); u16 path.
__device__ inline v16bf load_frag_a_scaled(const unsigned short* p, int ld, const float* sc) {
  int lane = threadIdx.x & 31;
  int m = lane & 15, half = lane >> 4;
  FragU f;
#pragma unroll
  for (int v = 0; v < 8; ++v) {
    int kb = 2 * v + 16 * half;
    f.u[2 * v]     = f2bf(bf2f(p[m * ld + kb])     * sc[kb]);
    f.u[2 * v + 1] = f2bf(bf2f(p[m * ld + kb + 1]) * sc[kb + 1]);
  }
  return f.v;
}

// B fragment from fragment-major LDS: lane l owns 8 consecutive dwords at
// base + l*8 (dword v = { B[2v+16*half][n], B[2v+16*half+1][n] }, l=(n&15)+16*half).
// Two ds_load_b128 per fragment.
__device__ inline v16bf load_frag_bf(const unsigned int* base) {
  int lane = threadIdx.x & 31;
  const uint4* q = (const uint4*)(base + lane * 8);
  FragU f;
  f.q[0] = q[0];
  f.q[1] = q[1];
  return f.v;
}

// Pack two B-tile rows (k even/odd) of 2 adjacent columns into pair-dwords.
__device__ inline void pack_pairs(const unsigned int drow[8],
                                  unsigned int c0[4], unsigned int c1[4]) {
#pragma unroll
  for (int pi = 0; pi < 4; ++pi) {
    unsigned int lo = drow[2 * pi], hi = drow[2 * pi + 1];
    c0[pi] = (lo & 0xffffu) | (hi << 16);
    c1[pi] = (lo >> 16) | (hi & 0xffff0000u);
  }
}

// ---------------------------------------------------------------- cast kernel
__global__ __launch_bounds__(256) void cast_f32_bf16_kernel(
    const float* __restrict__ in, unsigned short* __restrict__ out, int n4) {
  int i = blockIdx.x * blockDim.x + threadIdx.x;
  if (i >= n4) return;
  float4 f = ((const float4*)in)[i];
  unsigned long long p = (unsigned long long)f2bf(f.x)
                       | ((unsigned long long)f2bf(f.y) << 16)
                       | ((unsigned long long)f2bf(f.z) << 32)
                       | ((unsigned long long)f2bf(f.w) << 48);
  ((unsigned long long*)out)[i] = p;
}

// ----------------------------------------------------- generic bf16 GEMM core
// 128x128 block tile, BK=32, 256 threads (8 waves), each wave: 2x4 WMMA tiles.
#define EPI_SILU_BF16 0
#define EPI_BF16      1
#define EPI_GATE      2
#define EPI_F32       3

template <int EPI>
__global__ __launch_bounds__(256) void gemm_bf16_kernel(
    const unsigned short* __restrict__ A,   // [M,K] bf16 row-major
    const unsigned short* __restrict__ Bm,  // [K,N] bf16 row-major
    unsigned short* __restrict__ outb,      // bf16 output (SILU/BF16)
    float* __restrict__ outf,               // f32 output (F32)
    float* __restrict__ gio,                // in/out f32 (GATE: o *= sigmoid(acc))
    int M, int N, int K) {
  __shared__ __attribute__((aligned(16))) unsigned short As[128 * 40];  // row-major A
  __shared__ __attribute__((aligned(16))) unsigned int   Bs[2048];      // frag-major B (32x128)

  int n0 = blockIdx.x * 128;
  int m0 = blockIdx.y * 128;
  int t = threadIdx.x;
  int wave = t >> 5, lane = t & 31;
  int wm = wave >> 1, wn = wave & 1;        // 4x2 wave grid over (M,N)
  int nn = lane & 15, lh = lane >> 4;

  v8f acc[2][4];
#pragma unroll
  for (int i = 0; i < 2; ++i)
#pragma unroll
    for (int j = 0; j < 4; ++j) acc[i][j] = vzero8();

  int arow = t >> 1, acolg = (t & 1) * 16;
  int bn2 = (t & 63) * 2;          // column pair
  int bkb = (t >> 6) * 8;          // 8 k-rows (4 pairs)
  int bhalf = bkb >> 4;            // 0,0,1,1
  int bv0 = (bkb >> 1) & 7;        // 0,4,0,4
  int bg = bn2 >> 4;
  int bl0 = (bn2 & 15) + 16 * bhalf;

  for (int kk = 0; kk < K; kk += 32) {
    __syncthreads();
    {  // A tile 128x32 row-major
      const uint4* src = (const uint4*)(A + (size_t)(m0 + arow) * K + kk + acolg);
      uint4 v0 = src[0], v1 = src[1];
      *(uint4*)&As[arow * 40 + acolg]     = v0;
      *(uint4*)&As[arow * 40 + acolg + 8] = v1;
    }
    {  // B tile 32x128 -> fragment-major pack
      unsigned int drow[8];
#pragma unroll
      for (int r = 0; r < 8; ++r)
        drow[r] = *(const unsigned int*)(Bm + (size_t)(kk + bkb + r) * N + n0 + bn2);
      Pack4 c0, c1;
      pack_pairs(drow, c0.w, c1.w);
      *(uint4*)&Bs[bg * 256 + bl0 * 8 + bv0]       = c0.q;
      *(uint4*)&Bs[bg * 256 + (bl0 + 1) * 8 + bv0] = c1.q;
    }
    if (kk + 32 < K) {  // prefetch next K tiles (global_prefetch_b8)
      __builtin_prefetch(A + (size_t)(m0 + arow) * K + kk + 32 + acolg, 0, 1);
      __builtin_prefetch(Bm + (size_t)(kk + 32 + bkb) * N + n0 + bn2, 0, 1);
    }
    __syncthreads();
    v16bf a0 = load_frag_a128(&As[(wm * 32) * 40], 40);
    v16bf a1 = load_frag_a128(&As[(wm * 32 + 16) * 40], 40);
#pragma unroll
    for (int j = 0; j < 4; ++j) {
      v16bf bj = load_frag_bf(&Bs[(wn * 4 + j) * 256]);
      acc[0][j] = wmma_bf16(a0, bj, acc[0][j]);
      acc[1][j] = wmma_bf16(a1, bj, acc[1][j]);
    }
  }

#pragma unroll
  for (int i = 0; i < 2; ++i)
#pragma unroll
    for (int j = 0; j < 4; ++j)
#pragma unroll
      for (int r = 0; r < 8; ++r) {
        int rg = m0 + wm * 32 + i * 16 + r + 8 * lh;
        int cg = n0 + wn * 64 + j * 16 + nn;
        float v = acc[i][j][r];
        if (EPI == EPI_SILU_BF16) {
          float s = v / (1.0f + __expf(-v));
          outb[(size_t)rg * N + cg] = f2bf(s);
        } else if (EPI == EPI_BF16) {
          outb[(size_t)rg * N + cg] = f2bf(v);
        } else if (EPI == EPI_GATE) {
          float g = 1.0f / (1.0f + __expf(-v));
          size_t o = (size_t)rg * N + cg;
          gio[o] = gio[o] * g;
        } else {
          outf[(size_t)rg * N + cg] = v;
        }
      }
}

// -------------------------------------------------- chunked linear attention
// One block per (b, h, chunk). Computes:
//   attn = (q @ k^T) * Dm ; o_intra = attn @ v ; U = (k*dk)^T @ v
__global__ __launch_bounds__(256) void chunk_attn_kernel(
    const unsigned short* __restrict__ qh,  // [B,N,H,HD] bf16
    const unsigned short* __restrict__ kh,
    const unsigned short* __restrict__ vh,
    const float* __restrict__ log_decay,    // [H]
    float* __restrict__ o_out,              // [B,N,D] f32 (intra written)
    float* __restrict__ U) {                // [B,H,NC,HD,HD] f32
  __shared__ __attribute__((aligned(16))) unsigned short sQ[C_ * LDA_];   // q / attn row-major
  __shared__ __attribute__((aligned(16))) unsigned short sKT[HD_ * LDA_]; // k^T row-major [d][j]
  __shared__ __attribute__((aligned(16))) unsigned int   sKTp[4 * 2048];  // k^T frag-major (B for attn)
  __shared__ __attribute__((aligned(16))) unsigned int   sVp[4 * 2048];   // v frag-major (B role)
  __shared__ float sDk[C_];

  int idx = blockIdx.x;
  int c = idx % NC_;
  int h = (idx / NC_) % H_;
  int b = idx / (NC_ * H_);
  float ld = log_decay[h];

  int t = threadIdx.x;
  int wave = t >> 5, lane = t & 31;
  int wm = wave >> 1, wn = wave & 1;
  int nn = lane & 15, lh = lane >> 4;

  // ---- stage q (row-major) and k (row-major k^T + frag-major pair-packed)
#pragma unroll
  for (int rep = 0; rep < 8; ++rep) {
    int row = (t >> 4) + rep * 16;      // 0..127
    int colg = (t & 15) * 8;            // 0..120
    size_t gbase = ((size_t)(b * N_ + c * C_ + row) * H_ + h) * HD_ + colg;
    uint4 qv = *(const uint4*)(qh + gbase);
    *(uint4*)&sQ[row * LDA_ + colg] = qv;
    uint4 kv = *(const uint4*)(kh + gbase);
    const unsigned short* kp = (const unsigned short*)&kv;
#pragma unroll
    for (int e = 0; e < 8; ++e) sKT[(colg + e) * LDA_ + row] = kp[e];
    // kv's 4 dwords are the 4 consecutive (d,d+1) pairs p = colg/2 .. +3, col j=row
    int s    = colg >> 5;             // K-subtile (d block of 32)
    int half = (colg >> 4) & 1;
    int v0   = (colg >> 1) & 7;       // 0 or 4
    int g    = row >> 4;
    int l    = (row & 15) + 16 * half;
    *(uint4*)&sKTp[s * 2048 + g * 256 + l * 8 + v0] = kv;
  }
  // ---- stage v fragment-major (pairs over j)
  {
    int n2 = (t & 63) * 2;
#pragma unroll
    for (int rep = 0; rep < 4; ++rep) {
      int jb = (t >> 6) * 8 + rep * 32;  // 8-row block of j
      unsigned int drow[8];
#pragma unroll
      for (int r = 0; r < 8; ++r)
        drow[r] = *(const unsigned int*)(
            vh + ((size_t)(b * N_ + c * C_ + jb + r) * H_ + h) * HD_ + n2);
      Pack4 c0, c1;
      pack_pairs(drow, c0.w, c1.w);
      int s    = jb >> 5;
      int half = (jb >> 4) & 1;
      int v0   = (jb >> 1) & 7;
      int g    = n2 >> 4;
      int l0   = (n2 & 15) + 16 * half;
      *(uint4*)&sVp[s * 2048 + g * 256 + l0 * 8 + v0]       = c0.q;
      *(uint4*)&sVp[s * 2048 + g * 256 + (l0 + 1) * 8 + v0] = c1.q;
    }
  }
  if (t < C_) sDk[t] = __expf(ld * (float)(C_ - 1 - t));
  __syncthreads();

  v8f acc[2][4];

  // ---- GEMM1: attn = q @ k^T  (M=C, N=C, K=HD)
#pragma unroll
  for (int i = 0; i < 2; ++i)
#pragma unroll
    for (int j = 0; j < 4; ++j) acc[i][j] = vzero8();
  for (int kk = 0; kk < HD_; kk += 32) {
    v16bf a0 = load_frag_a128(&sQ[(wm * 32) * LDA_ + kk], LDA_);
    v16bf a1 = load_frag_a128(&sQ[(wm * 32 + 16) * LDA_ + kk], LDA_);
#pragma unroll
    for (int j = 0; j < 4; ++j) {
      v16bf bj = load_frag_bf(&sKTp[(kk >> 5) * 2048 + (wn * 4 + j) * 256]);
      acc[0][j] = wmma_bf16(a0, bj, acc[0][j]);
      acc[1][j] = wmma_bf16(a1, bj, acc[1][j]);
    }
  }
  __syncthreads();  // all waves done reading sQ
  // apply causal decay mask, convert to bf16, stash into sQ
#pragma unroll
  for (int i = 0; i < 2; ++i)
#pragma unroll
    for (int j = 0; j < 4; ++j)
#pragma unroll
      for (int r = 0; r < 8; ++r) {
        int ii = wm * 32 + i * 16 + r + 8 * lh;
        int jj = wn * 64 + j * 16 + nn;
        int diff = ii - jj;
        float val = (diff >= 0) ? acc[i][j][r] * __expf(ld * (float)diff) : 0.0f;
        sQ[ii * LDA_ + jj] = f2bf(val);
      }
  __syncthreads();

  // ---- GEMM2: o_intra = attn @ v  (M=C, N=HD, K=C)
#pragma unroll
  for (int i = 0; i < 2; ++i)
#pragma unroll
    for (int j = 0; j < 4; ++j) acc[i][j] = vzero8();
  for (int kk = 0; kk < C_; kk += 32) {
    v16bf a0 = load_frag_a128(&sQ[(wm * 32) * LDA_ + kk], LDA_);
    v16bf a1 = load_frag_a128(&sQ[(wm * 32 + 16) * LDA_ + kk], LDA_);
#pragma unroll
    for (int j = 0; j < 4; ++j) {
      v16bf bj = load_frag_bf(&sVp[(kk >> 5) * 2048 + (wn * 4 + j) * 256]);
      acc[0][j] = wmma_bf16(a0, bj, acc[0][j]);
      acc[1][j] = wmma_bf16(a1, bj, acc[1][j]);
    }
  }
#pragma unroll
  for (int i = 0; i < 2; ++i)
#pragma unroll
    for (int j = 0; j < 4; ++j)
#pragma unroll
      for (int r = 0; r < 8; ++r) {
        int ii = wm * 32 + i * 16 + r + 8 * lh;
        int ee = wn * 64 + j * 16 + nn;
        o_out[((size_t)(b * N_ + c * C_ + ii) * H_ + h) * HD_ + ee] = acc[i][j][r];
      }

  // ---- GEMM3: U = (k * dk)^T @ v  (M=HD, N=HD, K=C); A = sKT scaled by dk[j]
#pragma unroll
  for (int i = 0; i < 2; ++i)
#pragma unroll
    for (int j = 0; j < 4; ++j) acc[i][j] = vzero8();
  for (int kk = 0; kk < C_; kk += 32) {
    v16bf a0 = load_frag_a_scaled(&sKT[(wm * 32) * LDA_ + kk], LDA_, &sDk[kk]);
    v16bf a1 = load_frag_a_scaled(&sKT[(wm * 32 + 16) * LDA_ + kk], LDA_, &sDk[kk]);
#pragma unroll
    for (int j = 0; j < 4; ++j) {
      v16bf bj = load_frag_bf(&sVp[(kk >> 5) * 2048 + (wn * 4 + j) * 256]);
      acc[0][j] = wmma_bf16(a0, bj, acc[0][j]);
      acc[1][j] = wmma_bf16(a1, bj, acc[1][j]);
    }
  }
  size_t ubase = ((size_t)(b * H_ + h) * NC_ + c) * (HD_ * HD_);
#pragma unroll
  for (int i = 0; i < 2; ++i)
#pragma unroll
    for (int j = 0; j < 4; ++j)
#pragma unroll
      for (int r = 0; r < 8; ++r) {
        int dd = wm * 32 + i * 16 + r + 8 * lh;
        int ee = wn * 64 + j * 16 + nn;
        U[ubase + (size_t)dd * HD_ + ee] = acc[i][j][r];
      }
}

// ------------------------------------------------------- sequential state scan
// In-place U -> S (state BEFORE each chunk). One block per (b,h).
__global__ __launch_bounds__(256) void state_scan_kernel(
    float* __restrict__ U, const float* __restrict__ log_decay) {
  int bh = blockIdx.x;
  int h = bh % H_;
  float dS = __expf(log_decay[h] * (float)C_);
  int t = threadIdx.x;
  float s[64];
#pragma unroll
  for (int e = 0; e < 64; ++e) s[e] = 0.0f;
  for (int c = 0; c < NC_; ++c) {
    float* base = U + ((size_t)bh * NC_ + c) * (HD_ * HD_);
#pragma unroll
    for (int e = 0; e < 64; ++e) {
      int idx = e * 256 + t;
      float u = base[idx];
      base[idx] = s[e];
      s[e] = s[e] * dS + u;
    }
  }
}

// ------------------------------------------------- inter-chunk: o += (q*dq)@S
__global__ __launch_bounds__(256) void inter_chunk_kernel(
    const unsigned short* __restrict__ qh,  // [B,N,H,HD] bf16
    const float* __restrict__ S,            // [B,H,NC,HD,HD] f32 (pre-chunk states)
    const float* __restrict__ log_decay,
    float* __restrict__ o_out) {            // [B,N,D] f32 (accumulate)
  __shared__ __attribute__((aligned(16))) unsigned short sA[C_ * LDA_];  // q*dq row-major
  __shared__ __attribute__((aligned(16))) unsigned int   sBp[4 * 2048];  // S bf16 frag-major

  int idx = blockIdx.x;
  int c = idx % NC_;
  int h = (idx / NC_) % H_;
  int b = idx / (NC_ * H_);
  float ld = log_decay[h];

  int t = threadIdx.x;
  int wave = t >> 5, lane = t & 31;
  int wm = wave >> 1, wn = wave & 1;
  int nn = lane & 15, lh = lane >> 4;

  // ---- stage q scaled by dq = exp(ld*(row+1)), row-major
#pragma unroll
  for (int rep = 0; rep < 8; ++rep) {
    int row = (t >> 4) + rep * 16;
    int colg = (t & 15) * 8;
    size_t gbase = ((size_t)(b * N_ + c * C_ + row) * H_ + h) * HD_ + colg;
    uint4 qv = *(const uint4*)(qh + gbase);
    const unsigned short* qp = (const unsigned short*)&qv;
    float dq = __expf(ld * (float)(row + 1));
    Pack4 p0, p1;
#pragma unroll
    for (int e = 0; e < 4; ++e) p0.us[e] = f2bf(bf2f(qp[e]) * dq);
#pragma unroll
    for (int e = 0; e < 4; ++e) p0.us[4 + e] = f2bf(bf2f(qp[4 + e]) * dq);
    (void)p1;
    *(uint4*)&sA[row * LDA_ + colg] = p0.q;
  }
  // ---- stage S (f32 -> bf16) fragment-major (pairs over d)
  {
    int n2 = (t & 63) * 2;
    size_t sb = ((size_t)(b * H_ + h) * NC_ + c) * (HD_ * HD_);
#pragma unroll
    for (int rep = 0; rep < 4; ++rep) {
      int db = (t >> 6) * 8 + rep * 32;
      float2 fr[8];
#pragma unroll
      for (int r = 0; r < 8; ++r)
        fr[r] = *(const float2*)(S + sb + (size_t)(db + r) * HD_ + n2);
      Pack4 c0, c1;
#pragma unroll
      for (int pi = 0; pi < 4; ++pi) {
        c0.w[pi] = (unsigned int)f2bf(fr[2 * pi].x) | ((unsigned int)f2bf(fr[2 * pi + 1].x) << 16);
        c1.w[pi] = (unsigned int)f2bf(fr[2 * pi].y) | ((unsigned int)f2bf(fr[2 * pi + 1].y) << 16);
      }
      int s    = db >> 5;
      int half = (db >> 4) & 1;
      int v0   = (db >> 1) & 7;
      int g    = n2 >> 4;
      int l0   = (n2 & 15) + 16 * half;
      *(uint4*)&sBp[s * 2048 + g * 256 + l0 * 8 + v0]       = c0.q;
      *(uint4*)&sBp[s * 2048 + g * 256 + (l0 + 1) * 8 + v0] = c1.q;
    }
  }
  __syncthreads();

  v8f acc[2][4];
#pragma unroll
  for (int i = 0; i < 2; ++i)
#pragma unroll
    for (int j = 0; j < 4; ++j) acc[i][j] = vzero8();
  for (int kk = 0; kk < HD_; kk += 32) {
    v16bf a0 = load_frag_a128(&sA[(wm * 32) * LDA_ + kk], LDA_);
    v16bf a1 = load_frag_a128(&sA[(wm * 32 + 16) * LDA_ + kk], LDA_);
#pragma unroll
    for (int j = 0; j < 4; ++j) {
      v16bf bj = load_frag_bf(&sBp[(kk >> 5) * 2048 + (wn * 4 + j) * 256]);
      acc[0][j] = wmma_bf16(a0, bj, acc[0][j]);
      acc[1][j] = wmma_bf16(a1, bj, acc[1][j]);
    }
  }
#pragma unroll
  for (int i = 0; i < 2; ++i)
#pragma unroll
    for (int j = 0; j < 4; ++j)
#pragma unroll
      for (int r = 0; r < 8; ++r) {
        int ii = wm * 32 + i * 16 + r + 8 * lh;
        int ee = wn * 64 + j * 16 + nn;
        size_t o = ((size_t)(b * N_ + c * C_ + ii) * H_ + h) * HD_ + ee;
        o_out[o] = o_out[o] + acc[i][j][r];
      }
}

// ----------------------------------------------------------------- rmsnorm
__global__ __launch_bounds__(256) void rmsnorm_kernel(
    const float* __restrict__ o, const float* __restrict__ w,
    unsigned short* __restrict__ on) {
  __shared__ float red[256];
  int r = blockIdx.x;
  int t = threadIdx.x;
  const float* row = o + (size_t)r * D_;
  float ss = 0.0f;
  for (int i = t; i < D_; i += 256) { float v = row[i]; ss += v * v; }
  red[t] = ss;
  __syncthreads();
  for (int s = 128; s > 0; s >>= 1) {
    if (t < s) red[t] += red[t + s];
    __syncthreads();
  }
  float scale = rsqrtf(red[0] / (float)D_ + 1e-6f);
  for (int i = t; i < D_; i += 256)
    on[(size_t)r * D_ + i] = f2bf(row[i] * scale * w[i]);
}

// ================================================================== launcher
extern "C" void kernel_launch(void* const* d_in, const int* in_sizes, int n_in,
                              void* d_out, int out_size, void* d_ws, size_t ws_size,
                              hipStream_t stream) {
  const float* x         = (const float*)d_in[0];
  const float* log_decay = (const float*)d_in[1];
  const float* Wq        = (const float*)d_in[2];
  const float* Wk        = (const float*)d_in[3];
  const float* Wv        = (const float*)d_in[4];
  const float* Wo        = (const float*)d_in[5];
  const float* Wg1       = (const float*)d_in[6];
  const float* Wg2       = (const float*)d_in[7];
  const float* norm_w    = (const float*)d_in[8];
  float* out = (float*)d_out;

  char* ws = (char*)d_ws;
  size_t off = 0;
  auto take = [&](size_t bytes) -> char* {
    char* p = ws + off;
    off = (off + bytes + 255) & ~(size_t)255;
    return p;
  };

  const size_t ND = (size_t)B_ * N_ * D_;       // 8388608
  const size_t DD = (size_t)D_ * D_;            // 4194304
  const size_t DHD = (size_t)D_ * HD_;          // 262144
  const size_t NHD = (size_t)B_ * N_ * HD_;     // 524288
  const size_t UELEMS = (size_t)B_ * H_ * NC_ * HD_ * HD_;  // 8388608

  unsigned short* XH  = (unsigned short*)take(2 * ND);
  unsigned short* WQH = (unsigned short*)take(2 * DD);
  unsigned short* WKH = (unsigned short*)take(2 * DD);
  unsigned short* WVH = (unsigned short*)take(2 * DD);
  unsigned short* WOH = (unsigned short*)take(2 * DD);
  unsigned short* WG1H = (unsigned short*)take(2 * DHD);
  unsigned short* WG2H = (unsigned short*)take(2 * DHD);
  unsigned short* QH  = (unsigned short*)take(2 * ND);
  unsigned short* KH  = (unsigned short*)take(2 * ND);
  unsigned short* VH  = (unsigned short*)take(2 * ND);
  unsigned short* G1H = (unsigned short*)take(2 * NHD);
  float* OBUF = (float*)take(4 * ND);
  float* UBUF = (float*)take(4 * UELEMS);
  unsigned short* ONH = (unsigned short*)take(2 * ND);
  (void)ws_size; (void)in_sizes; (void)n_in; (void)out_size;

  auto cast = [&](const float* src, unsigned short* dst, size_t n) {
    int n4 = (int)(n / 4);
    cast_f32_bf16_kernel<<<(n4 + 255) / 256, 256, 0, stream>>>(src, dst, n4);
  };

  // 0) fp32 -> bf16 conversions
  cast(x, XH, ND);
  cast(Wq, WQH, DD);
  cast(Wk, WKH, DD);
  cast(Wv, WVH, DD);
  cast(Wo, WOH, DD);
  cast(Wg1, WG1H, DHD);
  cast(Wg2, WG2H, DHD);

  // 1) projections: q,k,v = silu(x@W) ; g1 = x@Wg1
  dim3 gFull(D_ / 128, MTOT_ / 128);   // (16, 32)
  gemm_bf16_kernel<EPI_SILU_BF16><<<gFull, 256, 0, stream>>>(
      XH, WQH, QH, nullptr, nullptr, MTOT_, D_, D_);
  gemm_bf16_kernel<EPI_SILU_BF16><<<gFull, 256, 0, stream>>>(
      XH, WKH, KH, nullptr, nullptr, MTOT_, D_, D_);
  gemm_bf16_kernel<EPI_SILU_BF16><<<gFull, 256, 0, stream>>>(
      XH, WVH, VH, nullptr, nullptr, MTOT_, D_, D_);
  dim3 gG1(HD_ / 128, MTOT_ / 128);    // (1, 32)
  gemm_bf16_kernel<EPI_BF16><<<gG1, 256, 0, stream>>>(
      XH, WG1H, G1H, nullptr, nullptr, MTOT_, HD_, D_);

  // 2) intra-chunk attention + per-chunk KV stats
  chunk_attn_kernel<<<B_ * H_ * NC_, 256, 0, stream>>>(
      QH, KH, VH, log_decay, OBUF, UBUF);

  // 3) sequential state scan (U -> S, in place)
  state_scan_kernel<<<B_ * H_, 256, 0, stream>>>(UBUF, log_decay);

  // 4) inter-chunk contribution: o += (q*dq) @ S
  inter_chunk_kernel<<<B_ * H_ * NC_, 256, 0, stream>>>(
      QH, UBUF, log_decay, OBUF);

  // 5) gate: o *= sigmoid(g1 @ Wg2)
  gemm_bf16_kernel<EPI_GATE><<<gFull, 256, 0, stream>>>(
      G1H, WG2H, nullptr, nullptr, OBUF, MTOT_, D_, HD_);

  // 6) rmsnorm -> bf16
  rmsnorm_kernel<<<MTOT_, 256, 0, stream>>>(OBUF, norm_w, ONH);

  // 7) final projection: out = on @ Wo (f32 output)
  gemm_bf16_kernel<EPI_F32><<<gFull, 256, 0, stream>>>(
      ONH, WOH, nullptr, out, nullptr, MTOT_, D_, D_);
}
